// ModuleCorefPropHoi_24790551232689
// MI455X (gfx1250) — compile-verified, matrix operand order checked by reading, not run on hardware
//
#include <hip/hip_runtime.h>

// ---------------------------------------------------------------------------
// MI455X / gfx1250 implementation of ModuleCorefPropHoi.
// - Pair-MLP ([512*512,128]@[128,128], ~34 GFLOP over 4 calls) is fused: h
//   lives only in LDS per (8i x 16j) pair tile, GEMM on v_wmma_f32_16x16x32_f16,
//   @W3 folded into the epilogue via a 16-lane shuffle reduction.
// - Dense GEMMs: 64x64 WMMA tiles, A row-major / B pre-transposed so every
//   fragment load is two 16B b128 loads per lane, and global->LDS staging uses
//   CDNA5 GLOBAL_LOAD_ASYNC_TO_LDS_B128 with double-buffered LDS (ASYNCcnt
//   tracked, one s_wait_asynccnt + one barrier per 32-K chunk).
// ---------------------------------------------------------------------------

typedef __attribute__((ext_vector_type(16))) _Float16 v16h;
typedef __attribute__((ext_vector_type(8)))  _Float16 v8h;
typedef __attribute__((ext_vector_type(8)))  float    v8f;

#define WMMA_F16(a, b, c) \
  __builtin_amdgcn_wmma_f32_16x16x32_f16(false, (a), false, (b), (short)0, (c), false, false)

#define MS 512   // spans
#define DD 768   // span vec dim
#define HH 128   // pair hidden dim

// Async DMA of 16 bytes (per active lane) from global memory into LDS.
// VGLOBAL op GLOBAL_LOAD_ASYNC_TO_LDS_B128: global vaddr -> LDS address vdst;
// tracked by ASYNCcnt. LDS byte address = low 32 bits of the generic pointer
__device__ __forceinline__ void async_ld_b128(const void* g, const void* l) {
  const unsigned lds = (unsigned)(unsigned long long)l;
  asm volatile("global_load_async_to_lds_b128 %0, %1, off"
               :: "v"(lds), "v"(g) : "memory");
}
__device__ __forceinline__ void wait_async0() {
  asm volatile("s_wait_asynccnt 0x0" ::: "memory");
}

// Fragment loader for a 16x32 f16 operand tile stored row-major with leading
// dim `ld` (rows = M for A, rows = N for transposed B). CDNA5 wave32 layout:
// lane L (0-15) -> row L, K halves {kh..kh+7, 16+kh..16+kh+7} with kh = 0;
// lanes 16-31 -> same rows, kh = 8. Two contiguous 8-half runs per lane.
__device__ __forceinline__ v16h frag_row_ld(const _Float16* src, int ld, int lane) {
  const int m  = lane & 15;
  const int kh = (lane >> 4) << 3;
  const _Float16* r = src + m * ld + kh;
  const v8h lo = *(const v8h*)(r);
  const v8h hi = *(const v8h*)(r + 16);
  return __builtin_shufflevector(lo, hi, 0, 1, 2, 3, 4, 5, 6, 7,
                                 8, 9, 10, 11, 12, 13, 14, 15);
}

// ---------------------------------------------------------------------------
// Generic C = act(A@B + bias) WMMA GEMM. A:[M,K] f16 row-major, BT:[N,K] f16
// (B transposed), C:[M,N] f32. Requires M%64==0, N%64==0, K%32==0 (true for
// every call site). 256 threads = 8 waves; block tile 64x64; each wave owns
// two 16x16 output tiles. Double-buffered async global->LDS staging.
// ---------------------------------------------------------------------------
__global__ __launch_bounds__(256) void gemm_f16f32(
    const _Float16* __restrict__ A, const _Float16* __restrict__ BT,
    const float* __restrict__ bias, float* __restrict__ C,
    int M, int N, int K, int act) {
  __shared__ _Float16 sA[2][64][40];    // 32 + 8 pad (80B rows -> 16B aligned)
  __shared__ _Float16 sBT[2][64][40];

  const int tid  = threadIdx.x;
  const int lane = tid & 31;
  const int wave = tid >> 5;
  const int bm = blockIdx.y * 64, bn = blockIdx.x * 64;
  const int wr = (wave >> 2) * 16, wc = (wave & 3) * 16;

  // Per-thread staging slot: 64 rows x 4 segments of 8 halves.
  const int r = tid >> 2, c = (tid & 3) * 8;
  const _Float16* gA  = A  + (size_t)(bm + r) * K + c;
  const _Float16* gBT = BT + (size_t)(bn + r) * K + c;

  // Prologue: asynchronously stage K-chunk 0 into buffer 0.
  async_ld_b128(gA,  &sA[0][r][c]);
  async_ld_b128(gBT, &sBT[0][r][c]);

  v8f acc0 = {}, acc1 = {};
  int buf = 0;
  for (int k0 = 0; k0 < K; k0 += 32) {
    wait_async0();        // this wave's async stores for chunk k0 are in LDS
    __syncthreads();      // all waves' chunk-k0 data visible; prior reads done
    if (k0 + 32 < K) {    // pipeline: stage the next chunk into the other buf
      async_ld_b128(gA  + k0 + 32, &sA[buf ^ 1][r][c]);
      async_ld_b128(gBT + k0 + 32, &sBT[buf ^ 1][r][c]);
    }
    v16h a0 = frag_row_ld(&sA[buf][wr][0],      40, lane);
    v16h a1 = frag_row_ld(&sA[buf][wr + 32][0], 40, lane);
    v16h bf = frag_row_ld(&sBT[buf][wc][0],     40, lane);
    acc0 = WMMA_F16(a0, bf, acc0);
    acc1 = WMMA_F16(a1, bf, acc1);
    buf ^= 1;
  }

  const int n  = bn + wc + (lane & 15);
  const int mo = (lane >> 4) * 8;
  const float bv = bias ? bias[n] : 0.0f;
#pragma unroll
  for (int rr = 0; rr < 8; ++rr) {
    float x0 = acc0[rr] + bv;
    float x1 = acc1[rr] + bv;
    if (act == 1) { x0 = fmaxf(x0, 0.0f); x1 = fmaxf(x1, 0.0f); }
    C[(size_t)(bm + wr + mo + rr) * N + n]      = x0;
    C[(size_t)(bm + wr + 32 + mo + rr) * N + n] = x1;
  }
}

// ---------------------------------------------------------------------------
// Fused pair scorer: for an (8 i x 16 j) tile of span pairs, build
// h = relu(left_i + right_j + dist_emb[bucket(|bi-bj|)]) in LDS (f16, b128
// stores), run [128,128]@[128,128] on WMMA with B fragments streamed from the
// cache-hot transposed W2, relu+b2 and the @W3 dot in the epilogue, then add
// unary_i + unary_j + b3, zero the diagonal, and emit scores.
// ---------------------------------------------------------------------------
__global__ __launch_bounds__(256) void pair_score_kernel(
    const float* __restrict__ left, const float* __restrict__ right,
    const int* __restrict__ begin, const float* __restrict__ dist_emb,
    const _Float16* __restrict__ W2T, const float* __restrict__ b2,
    const float* __restrict__ W3, const float* __restrict__ b3,
    const float* __restrict__ unary, float* __restrict__ scores) {
  __shared__ float sL[8][HH];              // 4 KB
  __shared__ float sR[16][HH];             // 8 KB
  __shared__ float sD[10][HH];             // 5 KB
  __shared__ _Float16 sH[128][HH + 8];     // ~34 KB, 272B rows (16B aligned)
  __shared__ int sBI[8], sBJ[16];

  const int tid  = threadIdx.x;
  const int lane = tid & 31;
  const int wave = tid >> 5;
  const int I0 = blockIdx.y * 8;
  const int J0 = blockIdx.x * 16;

  // Async-stage left/right/dist tiles into LDS (4 B128 per thread + tail).
  float* sLf = (float*)sL;
  float* sRf = (float*)sR;
  float* sDf = (float*)sD;
  const int t4 = tid * 4;
  async_ld_b128(left  + (size_t)I0 * HH + t4, &sLf[t4]);            // 1024 f32
  async_ld_b128(right + (size_t)J0 * HH + t4, &sRf[t4]);            // 2048 f32
  async_ld_b128(right + (size_t)J0 * HH + 1024 + t4, &sRf[1024 + t4]);
  async_ld_b128(dist_emb + t4, &sDf[t4]);                           // 1280 f32
  if (tid < 64) async_ld_b128(dist_emb + 1024 + t4, &sDf[1024 + t4]);
  if (tid < 8) sBI[tid] = begin[I0 + tid];
  if (tid >= 32 && tid < 48) sBJ[tid - 32] = begin[J0 + tid - 32];
  wait_async0();
  __syncthreads();

  // Build h: pair p = il*16 + jl; 2 threads per pair (64 k-values each),
  // packed 8 halves at a time -> 16B LDS stores.
  {
    const int p  = tid >> 1;
    const int il = p >> 4, jl = p & 15;
    const int kh = (tid & 1) * 64;
    int d = sBI[il] - sBJ[jl];
    d = d < 0 ? -d : d;
    int bk;
    if (d < 5) bk = d;
    else { bk = 31 - __builtin_clz(d) + 3; if (bk > 9) bk = 9; }
#pragma unroll
    for (int k8 = 0; k8 < 64; k8 += 8) {
      v8h pack;
#pragma unroll
      for (int u = 0; u < 8; ++u) {
        const int k = kh + k8 + u;
        pack[u] = (_Float16)fmaxf(sL[il][k] + sR[jl][k] + sD[bk][k], 0.0f);
      }
      *(v8h*)&sH[p][kh + k8] = pack;
    }
  }
  __syncthreads();

  // Each wave owns one 16-row tile of h; loop over all 8 n-tiles of W2.
  const int row0 = wave * 16;
  v16h afr[4];
#pragma unroll
  for (int kk = 0; kk < 4; ++kk)
    afr[kk] = frag_row_ld(&sH[row0][kk * 32], HH + 8, lane);

  const int nlo = lane & 15;
  const int mo  = (lane >> 4) * 8;
  float sr[8] = {0.f, 0.f, 0.f, 0.f, 0.f, 0.f, 0.f, 0.f};
#pragma unroll
  for (int nt = 0; nt < 8; ++nt) {
    const int n0 = nt * 16;
    v8f acc = {};
#pragma unroll
    for (int kk = 0; kk < 4; ++kk) {
      v16h bf = frag_row_ld(W2T + (size_t)n0 * HH + kk * 32, HH, lane);
      acc = WMMA_F16(afr[kk], bf, acc);
    }
    const float b2v = b2[n0 + nlo];
    const float w3v = W3[n0 + nlo];
#pragma unroll
    for (int rr = 0; rr < 8; ++rr)
      sr[rr] += fmaxf(acc[rr] + b2v, 0.0f) * w3v;   // relu(h2) . W3, partial over n
  }
  // Reduce over the 16 lanes holding the same row (xor masks stay in-half).
#pragma unroll
  for (int off = 8; off > 0; off >>= 1)
#pragma unroll
    for (int rr = 0; rr < 8; ++rr) sr[rr] += __shfl_xor(sr[rr], off, 32);

  if (nlo == 0) {
    const float b3v = b3[0];
#pragma unroll
    for (int rr = 0; rr < 8; ++rr) {
      const int p = row0 + mo + rr;
      const int i = I0 + (p >> 4);
      const int j = J0 + (p & 15);
      const float v = (i == j) ? 0.0f : (sr[rr] + b3v + unary[i] + unary[j]);
      scores[(size_t)i * MS + j] = v;
    }
  }
}

// ---------------------------------------------------------------------------
// Row softmax over masked scores; emits f16 probs (WMMA A operand for ctxt).
// ---------------------------------------------------------------------------
__global__ __launch_bounds__(256) void softmax_rows(
    const float* __restrict__ scores, const float* __restrict__ tri,
    _Float16* __restrict__ probsH) {
  __shared__ float red[256];
  const int i = blockIdx.x, tid = threadIdx.x;
  const float* srow = scores + (size_t)i * MS;
  const float* trow = tri + (size_t)i * MS;
  float x0 = srow[tid]       - (1.0f - trow[tid])       * 1e23f;
  float x1 = srow[tid + 256] - (1.0f - trow[tid + 256]) * 1e23f;
  red[tid] = fmaxf(x0, x1);
  __syncthreads();
  for (int s = 128; s > 0; s >>= 1) {
    if (tid < s) red[tid] = fmaxf(red[tid], red[tid + s]);
    __syncthreads();
  }
  const float m = red[0];
  __syncthreads();
  const float e0 = __expf(x0 - m), e1 = __expf(x1 - m);
  red[tid] = e0 + e1;
  __syncthreads();
  for (int s = 128; s > 0; s >>= 1) {
    if (tid < s) red[tid] += red[tid + s];
    __syncthreads();
  }
  const float inv = 1.0f / red[0];
  probsH[(size_t)i * MS + tid]       = (_Float16)(e0 * inv);
  probsH[(size_t)i * MS + tid + 256] = (_Float16)(e1 * inv);
}

// ------------------------- small elementwise kernels -----------------------
// f32 [K,N] -> f16 transposed [N,K] (for WMMA B operands).
__global__ void cvt_f16_T(const float* __restrict__ s, _Float16* __restrict__ dT,
                          int K, int N) {
  const int i = blockIdx.x * 256 + threadIdx.x;   // i indexes dT flat [N*K]
  if (i < N * K) {
    const int n = i / K, k = i - n * K;
    dT[i] = (_Float16)s[(size_t)k * N + n];
  }
}

__global__ void init_update(const float* __restrict__ src, float* __restrict__ updF,
                            _Float16* __restrict__ updH, _Float16* __restrict__ updTH,
                            int n) {
  const int i = blockIdx.x * 256 + threadIdx.x;
  if (i < n) {
    const float v = src[i];
    const int m = i / DD, k = i - m * DD;
    updF[i] = v;
    updH[i] = (_Float16)v;
    updTH[(size_t)k * MS + m] = (_Float16)v;   // [DD][MS] shadow (B operand)
  }
}

__global__ void concat_f16(const float* __restrict__ upd, const float* __restrict__ ctxt,
                           _Float16* __restrict__ cat, int n) {
  const int i = blockIdx.x * 256 + threadIdx.x;
  if (i < n) {
    const int m = i / DD, k = i - m * DD;
    cat[(size_t)m * (2 * DD) + k]      = (_Float16)upd[i];
    cat[(size_t)m * (2 * DD) + DD + k] = (_Float16)ctxt[i];
  }
}

__global__ void gate_update(float* __restrict__ updF, _Float16* __restrict__ updH,
                            _Float16* __restrict__ updTH,
                            const float* __restrict__ ctxt, const float* __restrict__ graw,
                            int n) {
  const int i = blockIdx.x * 256 + threadIdx.x;
  if (i < n) {
    const float g = 1.0f / (1.0f + __expf(-graw[i]));
    const float v = g * updF[i] + (1.0f - g) * ctxt[i];
    const int m = i / DD, k = i - m * DD;
    updF[i] = v;
    updH[i] = (_Float16)v;
    updTH[(size_t)k * MS + m] = (_Float16)v;
  }
}

__global__ void unary_kernel(const float* __restrict__ upd, const float* __restrict__ Wp,
                             const float* __restrict__ bp, float* __restrict__ out) {
  const int m = blockIdx.x * 256 + threadIdx.x;
  if (m < MS) {
    float s = 0.0f;
    for (int k = 0; k < DD; ++k) s += upd[(size_t)m * DD + k] * Wp[k];
    out[m] = s + bp[0];
  }
}

__global__ void copy_f32v4(const float* __restrict__ s, float* __restrict__ d, int n4) {
  const int i = blockIdx.x * 256 + threadIdx.x;
  if (i < n4) ((float4*)d)[i] = ((const float4*)s)[i];
}

__global__ void scatter_rows(const float* __restrict__ upd, const int* __restrict__ idx,
                             const int* __restrict__ len, float* __restrict__ outCand) {
  const int m = blockIdx.x;
  if (m >= len[0]) return;
  const int row = idx[m];
  for (int k = threadIdx.x; k < DD; k += 256)
    outCand[(size_t)row * DD + k] = upd[(size_t)m * DD + k];
}

// ---------------------------------------------------------------------------
extern "C" void kernel_launch(void* const* d_in, const int* in_sizes, int n_in,
                              void* d_out, int out_size, void* d_ws, size_t ws_size,
                              hipStream_t stream) {
  (void)in_sizes; (void)n_in; (void)out_size; (void)ws_size;
  const float* cand        = (const float*)d_in[0];   // [1,4096,768]
  const float* span_vecs   = (const float*)d_in[1];   // [1,512,768]
  const int*   span_begin  = (const int*)d_in[2];     // [1,512]
  const float* tri         = (const float*)d_in[4];   // [1,512,512]
  const float* span_scores = (const float*)d_in[5];   // [1,512,1] -> unary[512]
  const int*   prune_idx   = (const int*)d_in[6];     // [1,512]
  const int*   span_len    = (const int*)d_in[7];     // [1]
  // d_in[8] = coref_prop (device scalar, fixed at 3 by setup; loop unrolled on host)
  const float* Wl = (const float*)d_in[9];
  const float* bl = (const float*)d_in[10];
  const float* Wr = (const float*)d_in[11];
  const float* br = (const float*)d_in[12];
  const float* dist_emb = (const float*)d_in[13];
  const float* W2 = (const float*)d_in[14];
  const float* b2 = (const float*)d_in[15];
  const float* W3 = (const float*)d_in[16];
  const float* b3 = (const float*)d_in[17];
  const float* Wg = (const float*)d_in[18];
  const float* bg = (const float*)d_in[19];
  const float* Wp = (const float*)d_in[20];
  const float* bp = (const float*)d_in[21];

  // Output layout: [cand 4096*768 | update 512*768 | scores 512*512]
  float* out      = (float*)d_out;
  float* outCand  = out;
  float* updF     = out + (size_t)4096 * DD;   // also the working update buffer
  float* scores   = updF + (size_t)MS * DD;

  // Workspace partition (~10 MB), 256B aligned slices.
  char* w = (char*)d_ws;
  auto alloc = [&](size_t bytes) {
    void* p = (void*)w;
    w += (bytes + 255) & ~(size_t)255;
    return p;
  };
  _Float16* WlT    = (_Float16*)alloc((size_t)HH * DD * 2);       // [H][D]
  _Float16* WrT    = (_Float16*)alloc((size_t)HH * DD * 2);       // [H][D]
  _Float16* W2T    = (_Float16*)alloc((size_t)HH * HH * 2);       // [H][H]
  _Float16* WgT    = (_Float16*)alloc((size_t)DD * 2 * DD * 2);   // [D][2D]
  _Float16* updH   = (_Float16*)alloc((size_t)MS * DD * 2);       // [M][D] (A op)
  _Float16* updTH  = (_Float16*)alloc((size_t)DD * MS * 2);       // [D][M] (B op)
  _Float16* probsH = (_Float16*)alloc((size_t)MS * MS * 2);
  _Float16* catH   = (_Float16*)alloc((size_t)MS * 2 * DD * 2);
  float*    leftB  = (float*)alloc((size_t)MS * HH * 4);
  float*    rightB = (float*)alloc((size_t)MS * HH * 4);
  float*    ctxt   = (float*)alloc((size_t)MS * DD * 4);
  float*    graw   = (float*)alloc((size_t)MS * DD * 4);
  float*    unaryB = (float*)alloc((size_t)MS * 4);

  auto grid1 = [](int n) { return dim3((n + 255) / 256); };

  // One-time f32 -> f16 transposed weight conversions.
  cvt_f16_T<<<grid1(DD * HH), 256, 0, stream>>>(Wl, WlT, DD, HH);        // [768,128]->[128][768]
  cvt_f16_T<<<grid1(DD * HH), 256, 0, stream>>>(Wr, WrT, DD, HH);
  cvt_f16_T<<<grid1(HH * HH), 256, 0, stream>>>(W2, W2T, HH, HH);
  cvt_f16_T<<<grid1(2 * DD * DD), 256, 0, stream>>>(Wg, WgT, 2 * DD, DD); // [1536,768]->[768][1536]

  // update = span_vecs (copy into output-region working buffer + f16 shadows).
  init_update<<<grid1(MS * DD), 256, 0, stream>>>(span_vecs, updF, updH, updTH, MS * DD);

  const dim3 gProj(HH / 64, MS / 64);   // (2, 8)
  const dim3 gDDn(DD / 64, MS / 64);    // (12, 8)
  const dim3 gPair(MS / 16, MS / 8);    // (32, 64)

  // scores = add_scores(ff_pairs(update), span_scores)
  gemm_f16f32<<<gProj, 256, 0, stream>>>(updH, WlT, bl, leftB,  MS, HH, DD, 0);
  gemm_f16f32<<<gProj, 256, 0, stream>>>(updH, WrT, br, rightB, MS, HH, DD, 0);
  pair_score_kernel<<<gPair, 256, 0, stream>>>(leftB, rightB, span_begin, dist_emb,
                                               W2T, b2, W3, b3, span_scores, scores);

  for (int it = 0; it < 3; ++it) {   // coref_prop == 3 (fixed by setup)
    softmax_rows<<<MS, 256, 0, stream>>>(scores, tri, probsH);
    // ctxt = probs @ update : A = probsH [M][M], BT = updTH [D][M] (old update)
    gemm_f16f32<<<gDDn, 256, 0, stream>>>(probsH, updTH, nullptr, ctxt, MS, DD, MS, 0);
    concat_f16<<<grid1(MS * DD), 256, 0, stream>>>(updF, ctxt, catH, MS * DD);
    gemm_f16f32<<<gDDn, 256, 0, stream>>>(catH, WgT, bg, graw, MS, DD, 2 * DD, 0);
    gate_update<<<grid1(MS * DD), 256, 0, stream>>>(updF, updH, updTH, ctxt, graw, MS * DD);
    unary_kernel<<<grid1(MS), 256, 0, stream>>>(updF, Wp, bp, unaryB);
    gemm_f16f32<<<gProj, 256, 0, stream>>>(updH, WlT, bl, leftB,  MS, HH, DD, 0);
    gemm_f16f32<<<gProj, 256, 0, stream>>>(updH, WrT, br, rightB, MS, HH, DD, 0);
    pair_score_kernel<<<gPair, 256, 0, stream>>>(leftB, rightB, span_begin, dist_emb,
                                                 W2T, b2, W3, b3, unaryB, scores);
  }

  // new_cand: copy cand, then overwrite first span_len rows at prune indices.
  copy_f32v4<<<grid1(4096 * DD / 4), 256, 0, stream>>>(cand, outCand, 4096 * DD / 4);
  scatter_rows<<<MS, 256, 0, stream>>>(updF, prune_idx, span_len, outCand);
}
